// MambaModel_31018253811835
// MI455X (gfx1250) — matrix operand order repaired and checked by reading, not compiled, not run on hardware
//
#include <hip/hip_runtime.h>

// Mamba forward for MI455X (gfx1250, wave32).
// GEMMs: 128x128x32 block tiles, double-buffered LDS filled by async
//        global->LDS DMA (global_load_async_to_lds_b128 / ASYNCcnt),
//        8 v_wmma_f32_16x16x32_f16 per wave per K-step (32x64 region/wave).
// SSM scan: chunked 2-pass parallel linear scan (32 chunks x 128 steps).

#define B_  2
#define L_  4096
#define DM  768
#define NL  4
#define DI  1536
#define DS  16
#define DC  4
#define DR  48
#define XW  (DR + 2*DS)     // 80, x_proj output width
#define NCHUNK 32
#define CHUNK  (L_/NCHUNK)  // 128

typedef __attribute__((ext_vector_type(16))) _Float16     v16h;
typedef __attribute__((ext_vector_type(8)))  float        v8f;
typedef __attribute__((ext_vector_type(8)))  unsigned int v8u;

__device__ __forceinline__ float sigmoidf_(float x) { return 1.0f / (1.0f + __expf(-x)); }
__device__ __forceinline__ float siluf_(float x)    { return x * sigmoidf_(x); }
__device__ __forceinline__ float softplusf_(float x){ return (x > 20.0f) ? x : log1pf(__expf(x)); }

// ---- CDNA5 async global->LDS copy (ASYNCcnt-tracked), inline asm per ISA 08_async_tensor ----
typedef __attribute__((address_space(3))) char lds_char;

__device__ __forceinline__ unsigned lds_off32(const void* p) {
  // generic -> LDS addrspace cast folds to the raw 32-bit LDS byte offset
  return (unsigned)(unsigned long long)(lds_char*)(char*)p;
}
__device__ __forceinline__ void async_load_b128(unsigned lds_off, const void* gaddr) {
  asm volatile("global_load_async_to_lds_b128 %0, %1, off"
               :: "v"(lds_off), "v"(gaddr) : "memory");
}
__device__ __forceinline__ void wait_async0() {
  asm volatile("s_wait_asynccnt 0" ::: "memory");
}
__device__ __forceinline__ void wait_ds0() {
  asm volatile("s_wait_dscnt 0" ::: "memory");
}

// ---------------- transpose [B, DM, L] -> [B, L, DM] (LDS tiled) ----------------
__global__ __launch_bounds__(256) void k_transpose(const float* __restrict__ src,
                                                   float* __restrict__ dst) {
  __shared__ float tile[32][33];
  int t0 = blockIdx.x * 32, d0 = blockIdx.y * 32, b = blockIdx.z;
  int tx = threadIdx.x, ty = threadIdx.y;          // block (32, 8)
#pragma unroll
  for (int j = 0; j < 4; j++) {
    int d = d0 + ty + 8 * j;
    tile[ty + 8 * j][tx] = src[((size_t)b * DM + d) * L_ + t0 + tx];
  }
  __syncthreads();
#pragma unroll
  for (int j = 0; j < 4; j++) {
    int t = t0 + ty + 8 * j;
    dst[((size_t)b * L_ + t) * DM + d0 + tx] = tile[tx][ty + 8 * j];
  }
}

// ---------------- f32 -> f16 cast (weights) ----------------
__global__ __launch_bounds__(256) void k_cast_h(const float* __restrict__ s,
                                                _Float16* __restrict__ d, int n) {
  for (int i = blockIdx.x * 256 + threadIdx.x; i < n; i += gridDim.x * 256)
    d[i] = (_Float16)s[i];
}

// ---------------- RMSNorm row -> f16 ----------------
__global__ __launch_bounds__(256) void k_rmsnorm_cast(const float* __restrict__ x,
                                                      const float* __restrict__ w,
                                                      _Float16* __restrict__ o) {
  __shared__ float red[256];
  size_t r = blockIdx.x;
  const float* xr = x + r * DM;
  int tid = threadIdx.x;
  float ss = 0.0f;
  for (int i = tid; i < DM; i += 256) { float v = xr[i]; ss += v * v; }
  red[tid] = ss; __syncthreads();
  for (int s = 128; s > 0; s >>= 1) { if (tid < s) red[tid] += red[tid + s]; __syncthreads(); }
  float rs = rsqrtf(red[0] / (float)DM + 1e-5f);
  for (int i = tid; i < DM; i += 256) o[r * DM + i] = (_Float16)(xr[i] * rs * w[i]);
}

// ---------------- WMMA GEMM: out[M,N] (f32) = A[M,K] (f16) * W[N,K]^T (f16) ----------------
// Block 256 thr = 8 waves; block tile 128x128x32; each wave computes a 32x64
// region (2 A-frags x 4 B-frags -> 8 wmma per K-step). Double-buffered LDS
// filled by async global->LDS DMA (4 b128 per thread per K-step).
__device__ __forceinline__ v16h fragAB(const _Float16* p0, const _Float16* p1) {
  uint4 a = *reinterpret_cast<const uint4*>(p0);
  uint4 b = *reinterpret_cast<const uint4*>(p1);
  v8u u;
  u[0] = a.x; u[1] = a.y; u[2] = a.z; u[3] = a.w;
  u[4] = b.x; u[5] = b.y; u[6] = b.z; u[7] = b.w;
  return __builtin_bit_cast(v16h, u);
}

__global__ __launch_bounds__(256) void k_gemm_f16(const _Float16* __restrict__ A,
                                                  const _Float16* __restrict__ W,
                                                  float* __restrict__ out,
                                                  int M, int N, int K, int accum) {
  // 48-half row stride: 96B (b128-aligned, bank-skewed); 2 buffers for ping-pong
  __shared__ __attribute__((aligned(16))) _Float16 sA[2][128][48];
  __shared__ __attribute__((aligned(16))) _Float16 sB[2][128][48];
  int tid  = threadIdx.x;
  int lane = tid & 31, wv = tid >> 5;
  int mBase = blockIdx.y * 128, nBase = blockIdx.x * 128;
  int mreg = (wv & 3) * 32;         // wave's 32-row region within the tile
  int nreg = (wv >> 2) * 64;        // wave's 64-col region within the tile
  v8f acc[2][4];
#pragma unroll
  for (int i = 0; i < 2; i++)
#pragma unroll
    for (int j = 0; j < 4; j++) acc[i][j] = (v8f){};

  // copy lanes: each thread moves rows r0 and r0+64 of both tiles, k-group g0
  int r0 = tid >> 2;                // 0..63
  int g0 = (tid & 3) * 8;           // 8-half (b128) k-subgroup

  // clamp OOB copy rows: garbage only reaches never-stored rows/cols
  int mA0 = mBase + r0;       if (mA0 >= M) mA0 = M - 1;
  int mA1 = mBase + r0 + 64;  if (mA1 >= M) mA1 = M - 1;
  int nB0 = nBase + r0;       if (nB0 >= N) nB0 = N - 1;
  int nB1 = nBase + r0 + 64;  if (nB1 >= N) nB1 = N - 1;
  const _Float16* gA0 = A + (size_t)mA0 * K + g0;
  const _Float16* gA1 = A + (size_t)mA1 * K + g0;
  const _Float16* gB0 = W + (size_t)nB0 * K + g0;
  const _Float16* gB1 = W + (size_t)nB1 * K + g0;

  unsigned lA0[2] = { lds_off32(&sA[0][r0][g0]),      lds_off32(&sA[1][r0][g0]) };
  unsigned lA1[2] = { lds_off32(&sA[0][r0 + 64][g0]), lds_off32(&sA[1][r0 + 64][g0]) };
  unsigned lB0[2] = { lds_off32(&sB[0][r0][g0]),      lds_off32(&sB[1][r0][g0]) };
  unsigned lB1[2] = { lds_off32(&sB[0][r0 + 64][g0]), lds_off32(&sB[1][r0 + 64][g0]) };

  int nk = K >> 5;                  // K multiple of 32 for all uses
  // prologue: DMA tile 0
  async_load_b128(lA0[0], gA0);
  async_load_b128(lA1[0], gA1);
  async_load_b128(lB0[0], gB0);
  async_load_b128(lB1[0], gB1);

  int ml  = lane & 15;
  int kbA = (lane < 16) ? 0 : 8;
  int kbB = (lane >> 4) * 8;

  for (int i = 0; i < nk; i++) {
    wait_async0();            // own async writes done
    __syncthreads();          // everyone's async writes done -> buf[i&1] ready
    if (i + 1 < nk) {         // DMA next tile into the other buffer
      size_t adv = (size_t)(i + 1) * 32;
      int nb = (i + 1) & 1;
      async_load_b128(lA0[nb], gA0 + adv);
      async_load_b128(lA1[nb], gA1 + adv);
      async_load_b128(lB0[nb], gB0 + adv);
      async_load_b128(lB1[nb], gB1 + adv);
    }
    int buf = i & 1;
    // A fragments: lanes 0-15 K {0..7,16..23}, lanes 16-31 K {8..15,24..31}, M = lane%16
    v16h a0 = fragAB(&sA[buf][mreg + ml][kbA],      &sA[buf][mreg + ml][kbA + 16]);
    v16h a1 = fragAB(&sA[buf][mreg + 16 + ml][kbA], &sA[buf][mreg + 16 + ml][kbA + 16]);
#pragma unroll
    for (int j = 0; j < 4; j++) {
      // B fragment: N = lane%16, lanes 0-15 K {0..7,16..23}, lanes 16-31 K {8..15,24..31}
      const _Float16* bj = &sB[buf][nreg + j * 16 + ml][kbB];
      v16h bf = fragAB(bj, bj + 16);
      acc[0][j] = __builtin_amdgcn_wmma_f32_16x16x32_f16(false, a0, false, bf, (short)0, acc[0][j], false, false);
      acc[1][j] = __builtin_amdgcn_wmma_f32_16x16x32_f16(false, a1, false, bf, (short)0, acc[1][j], false, false);
    }
    wait_ds0();               // no LDS read outstanding when this buf is re-filled
  }

  // C/D layout: VGPR r -> M = r + (lane/16)*8, N = lane%16
  int mloc = (lane >> 4) * 8, nloc = lane & 15;
#pragma unroll
  for (int mt = 0; mt < 2; mt++) {
#pragma unroll
    for (int j = 0; j < 4; j++) {
      int colc = nBase + nreg + j * 16 + nloc;
      if (colc >= N) continue;
#pragma unroll
      for (int r = 0; r < 8; r++) {
        int row = mBase + mreg + mt * 16 + mloc + r;
        if (row >= M) continue;
        size_t o = (size_t)row * N + colc;
        out[o] = accum ? out[o] + acc[mt][j][r] : acc[mt][j][r];
      }
    }
  }
}

// ---------------- causal depthwise conv (width 4) + SiLU ----------------
__global__ __launch_bounds__(256) void k_conv_silu(const float* __restrict__ xz,
                                                   const float* __restrict__ cw,
                                                   const float* __restrict__ cb,
                                                   float* __restrict__ u,
                                                   _Float16* __restrict__ uh) {
  int n = B_ * L_ * DI;
  for (int i = blockIdx.x * 256 + threadIdx.x; i < n; i += gridDim.x * 256) {
    int d = i % DI; int row = i / DI; int t = row % L_;
    int rb = row - t;   // b*L_
    float s = cb[d];
#pragma unroll
    for (int j = 0; j < DC; j++) {
      int tt = t - (DC - 1) + j;
      if (tt >= 0) s += cw[d * DC + j] * xz[((size_t)(rb + tt)) * (2 * DI) + d];
    }
    float v = siluf_(s);
    u[i] = v; uh[i] = (_Float16)v;
  }
}

// ---------------- dt = softplus(dtr @ dt_proj_w^T + b) ----------------
__global__ __launch_bounds__(256) void k_dtproj(const float* __restrict__ xdb,
                                                const float* __restrict__ w,
                                                const float* __restrict__ bias,
                                                float* __restrict__ dt) {
  int n = B_ * L_ * DI;
  for (int i = blockIdx.x * 256 + threadIdx.x; i < n; i += gridDim.x * 256) {
    int d = i % DI; size_t row = (size_t)(i / DI);
    const float* dr = xdb + row * XW;
    const float* wr = w + d * DR;
    float s = bias[d];
#pragma unroll
    for (int j = 0; j < DR; j++) s += dr[j] * wr[j];
    dt[i] = softplusf_(s);
  }
}

// ---------------- scan pass 1: per-chunk h_end and cumulative decay ----------------
__global__ __launch_bounds__(256) void k_scan_pass1(const float* __restrict__ dt,
                                                    const float* __restrict__ u,
                                                    const float* __restrict__ xdb,
                                                    const float* __restrict__ A_log,
                                                    float* __restrict__ hend,
                                                    float* __restrict__ cumA) {
  int idx = blockIdx.x * 256 + threadIdx.x;        // = b*NCHUNK*DI + c*DI + d
  int d = idx % DI; int c = (idx / DI) % NCHUNK; int b = idx / (DI * NCHUNK);
  float a[DS], h[DS], p[DS];
#pragma unroll
  for (int s = 0; s < DS; s++) { a[s] = -__expf(A_log[d * DS + s]); h[s] = 0.0f; p[s] = 1.0f; }
  int t0 = c * CHUNK;
  for (int t = t0; t < t0 + CHUNK; t++) {
    size_t row = (size_t)b * L_ + t;
    __builtin_prefetch(&dt[(row + 8) * DI + d], 0, 1);
    __builtin_prefetch(&u[(row + 8) * DI + d], 0, 1);
    float dtv = dt[row * DI + d], uv = u[row * DI + d];
    const float* bp = xdb + row * XW + DR;
    float du = dtv * uv;
#pragma unroll
    for (int s = 0; s < DS; s++) {
      float e = __expf(dtv * a[s]);
      h[s] = h[s] * e + du * bp[s];
      p[s] *= e;
    }
  }
  size_t o = (size_t)idx * DS;
#pragma unroll
  for (int s = 0; s < DS; s++) { hend[o + s] = h[s]; cumA[o + s] = p[s]; }
}

// ---------------- scan combine: chain carries across chunks (hend := carry-in) ----------------
__global__ __launch_bounds__(256) void k_scan_combine(float* __restrict__ hend,
                                                      const float* __restrict__ cumA) {
  int idx = blockIdx.x * 256 + threadIdx.x;        // = b*DI*DS + d*DS + s
  int s = idx % DS; int d = (idx / DS) % DI; int b = idx / (DS * DI);
  float carry = 0.0f;
  for (int c = 0; c < NCHUNK; c++) {
    size_t o = (((size_t)b * NCHUNK + c) * DI + d) * DS + s;
    float he = hend[o], pa = cumA[o];
    hend[o] = carry;                 // becomes carry-in for pass 2
    carry = he + pa * carry;
  }
}

// ---------------- scan pass 2: exact re-scan from carry-in, emit gated y (f16) ----------------
__global__ __launch_bounds__(256) void k_scan_pass2(const float* __restrict__ dt,
                                                    const float* __restrict__ u,
                                                    const float* __restrict__ xdb,
                                                    const float* __restrict__ xz,
                                                    const float* __restrict__ A_log,
                                                    const float* __restrict__ Dw,
                                                    const float* __restrict__ carry,
                                                    _Float16* __restrict__ yh) {
  int idx = blockIdx.x * 256 + threadIdx.x;
  int d = idx % DI; int c = (idx / DI) % NCHUNK; int b = idx / (DI * NCHUNK);
  float a[DS], h[DS];
  size_t co = (size_t)idx * DS;
#pragma unroll
  for (int s = 0; s < DS; s++) { a[s] = -__expf(A_log[d * DS + s]); h[s] = carry[co + s]; }
  float Dv = Dw[d];
  int t0 = c * CHUNK;
  for (int t = t0; t < t0 + CHUNK; t++) {
    size_t row = (size_t)b * L_ + t;
    __builtin_prefetch(&dt[(row + 8) * DI + d], 0, 1);
    __builtin_prefetch(&u[(row + 8) * DI + d], 0, 1);
    float dtv = dt[row * DI + d], uv = u[row * DI + d];
    const float* bp = xdb + row * XW + DR;
    const float* cp = xdb + row * XW + DR + DS;
    float du = dtv * uv;
    float y = 0.0f;
#pragma unroll
    for (int s = 0; s < DS; s++) {
      float e = __expf(dtv * a[s]);
      h[s] = h[s] * e + du * bp[s];
      y += h[s] * cp[s];
    }
    float zv = xz[row * (2 * DI) + DI + d];
    float outv = (y + uv * Dv) * siluf_(zv);
    yh[row * DI + d] = (_Float16)outv;
  }
}

// ---------------- final RMSNorm of last valid token per batch row ----------------
__global__ __launch_bounds__(256) void k_final_norm(const float* __restrict__ x,
                                                    const unsigned char* __restrict__ mask,
                                                    const float* __restrict__ nfw,
                                                    float* __restrict__ out) {
  __shared__ float red[256];
  __shared__ int sidx;
  int b = blockIdx.x, tid = threadIdx.x;
  int cnt = 0;
  for (int i = tid; i < L_; i += 256) cnt += mask[(size_t)b * L_ + i] ? 1 : 0;
  red[tid] = (float)cnt; __syncthreads();
  for (int s = 128; s > 0; s >>= 1) { if (tid < s) red[tid] += red[tid + s]; __syncthreads(); }
  if (tid == 0) sidx = (int)red[0] - 1;
  __syncthreads();
  const float* row = x + ((size_t)b * L_ + sidx) * DM;
  float ss = 0.0f;
  for (int i = tid; i < DM; i += 256) { float v = row[i]; ss += v * v; }
  red[tid] = ss; __syncthreads();
  for (int s = 128; s > 0; s >>= 1) { if (tid < s) red[tid] += red[tid + s]; __syncthreads(); }
  float rs = rsqrtf(red[0] / (float)DM + 1e-5f);
  for (int i = tid; i < DM; i += 256) out[b * DM + i] = row[i] * rs * nfw[i];
}

// ---------------- host-side orchestration ----------------
extern "C" void kernel_launch(void* const* d_in, const int* in_sizes, int n_in,
                              void* d_out, int out_size, void* d_ws, size_t ws_size,
                              hipStream_t stream) {
  const float* batch          = (const float*)d_in[0];
  const unsigned char* mask   = (const unsigned char*)d_in[1];   // jnp.bool_ -> 1 byte/elt
  const float* in_proj_w      = (const float*)d_in[2];
  const float* conv_w         = (const float*)d_in[3];
  const float* conv_b         = (const float*)d_in[4];
  const float* x_proj_w       = (const float*)d_in[5];
  const float* dt_proj_w      = (const float*)d_in[6];
  const float* dt_proj_b      = (const float*)d_in[7];
  const float* A_log          = (const float*)d_in[8];
  const float* Dvec           = (const float*)d_in[9];
  const float* out_proj_w     = (const float*)d_in[10];
  const float* norm_w         = (const float*)d_in[11];
  const float* norm_f_w       = (const float*)d_in[12];
  (void)in_sizes; (void)n_in; (void)out_size; (void)ws_size;

  char* base = (char*)d_ws;
  size_t off = 0;
  auto carve = [&](size_t bytes) -> void* {
    void* p = base + off;
    off += (bytes + 255) & ~(size_t)255;
    return p;
  };
  float*    x    = (float*)   carve((size_t)B_ * L_ * DM * 4);
  _Float16* hn   = (_Float16*)carve((size_t)B_ * L_ * DM * 2);
  float*    xz   = (float*)   carve((size_t)B_ * L_ * 2 * DI * 4);
  float*    u    = (float*)   carve((size_t)B_ * L_ * DI * 4);
  _Float16* uh   = (_Float16*)carve((size_t)B_ * L_ * DI * 2);
  float*    xdb  = (float*)   carve((size_t)B_ * L_ * XW * 4);
  float*    dt   = (float*)   carve((size_t)B_ * L_ * DI * 4);
  _Float16* yh   = (_Float16*)carve((size_t)B_ * L_ * DI * 2);
  float*    hend = (float*)   carve((size_t)B_ * NCHUNK * DI * DS * 4);
  float*    cumA = (float*)   carve((size_t)B_ * NCHUNK * DI * DS * 4);
  _Float16* wIn  = (_Float16*)carve((size_t)2 * DI * DM * 2);
  _Float16* wXp  = (_Float16*)carve((size_t)XW * DI * 2);
  _Float16* wOp  = (_Float16*)carve((size_t)DM * DI * 2);

  const int rows = B_ * L_;   // 8192

  k_transpose<<<dim3(L_ / 32, DM / 32, B_), dim3(32, 8), 0, stream>>>(batch, x);

  for (int l = 0; l < NL; l++) {
    const float* ipw = in_proj_w  + (size_t)l * 2 * DI * DM;
    const float* xpw = x_proj_w   + (size_t)l * XW * DI;
    const float* opw = out_proj_w + (size_t)l * DM * DI;
    const float* alg = A_log      + (size_t)l * DI * DS;

    k_cast_h<<<1024, 256, 0, stream>>>(ipw, wIn, 2 * DI * DM);
    k_cast_h<<<64,   256, 0, stream>>>(xpw, wXp, XW * DI);
    k_cast_h<<<512,  256, 0, stream>>>(opw, wOp, DM * DI);

    k_rmsnorm_cast<<<rows, 256, 0, stream>>>(x, norm_w + (size_t)l * DM, hn);

    k_gemm_f16<<<dim3((2 * DI + 127) / 128, rows / 128), 256, 0, stream>>>(
        hn, wIn, xz, rows, 2 * DI, DM, 0);

    k_conv_silu<<<2048, 256, 0, stream>>>(
        xz, conv_w + (size_t)l * DI * DC, conv_b + (size_t)l * DI, u, uh);

    k_gemm_f16<<<dim3((XW + 127) / 128, rows / 128), 256, 0, stream>>>(
        uh, wXp, xdb, rows, XW, DI, 0);

    k_dtproj<<<2048, 256, 0, stream>>>(
        xdb, dt_proj_w + (size_t)l * DI * DR, dt_proj_b + (size_t)l * DI, dt);

    k_scan_pass1<<<(B_ * NCHUNK * DI) / 256, 256, 0, stream>>>(dt, u, xdb, alg, hend, cumA);
    k_scan_combine<<<(B_ * DI * DS) / 256, 256, 0, stream>>>(hend, cumA);
    k_scan_pass2<<<(B_ * NCHUNK * DI) / 256, 256, 0, stream>>>(
        dt, u, xdb, xz, alg, Dvec + (size_t)l * DI, hend, yh);

    k_gemm_f16<<<dim3((DM + 127) / 128, rows / 128), 256, 0, stream>>>(
        yh, wOp, x, rows, DM, DI, 1);
  }

  k_final_norm<<<B_, 256, 0, stream>>>(x, mask, norm_f_w, (float*)d_out);
}
